// Qwen3Attention_12438225289547
// MI455X (gfx1250) — compile-verified, matrix operand order checked by reading, not run on hardware
//
#include <hip/hip_runtime.h>
#include <hip/hip_bf16.h>

typedef __attribute__((ext_vector_type(16))) __bf16 v16bf;
typedef __attribute__((ext_vector_type(8)))  __bf16 v8bf;
typedef __attribute__((ext_vector_type(8)))  float  v8f;

static constexpr int BATCH = 2;
static constexpr int SEQ   = 2048;
static constexpr int HIDN  = 2048;
static constexpr int NH    = 16;
static constexpr int HD    = 128;
static constexpr int WIN   = 1024;
static constexpr int BS    = BATCH * SEQ;   // 4096 rows
static constexpr float EPSV  = 1e-6f;
static constexpr float THETA = 10000.0f;

// ---------------------------------------------------------------------------
// helpers
// ---------------------------------------------------------------------------
__device__ inline v16bf cat8(v8bf lo, v8bf hi) {
    v16bf r;
#pragma unroll
    for (int e = 0; e < 8; ++e) { r[e] = lo[e]; r[e + 8] = hi[e]; }
    return r;
}
__device__ inline v16bf load_frag(const __bf16* p) {
    // fragment layout (16-bit A/B, wave32): lane-half h owns k = 8h..8h+7 and 16+8h..16+8h+7
    return cat8(*(const v8bf*)p, *(const v8bf*)(p + 16));
}

// ---------------------------------------------------------------------------
// f32 -> bf16 conversion
// ---------------------------------------------------------------------------
__global__ __launch_bounds__(256) void cvt_f32_bf16(const float* __restrict__ src,
                                                    __bf16* __restrict__ dst, int n) {
    int i = (blockIdx.x * 256 + threadIdx.x) * 4;
    if (i + 3 < n) {
        float4 f = *(const float4*)(src + i);
        dst[i + 0] = (__bf16)f.x;
        dst[i + 1] = (__bf16)f.y;
        dst[i + 2] = (__bf16)f.z;
        dst[i + 3] = (__bf16)f.w;
    }
}

// ---------------------------------------------------------------------------
// bf16 WMMA GEMM:  C[M,N] = A[M,K] * Bw[N,K]^T   (both K-contiguous)
// 128x128 block tile, 8 waves (4 along M x 2 along N), each wave 32x64.
// mode 0: bf16 row-major out;  mode 1: f32 row-major out;
// mode 2: bf16 "V-transposed" out -> [b, n, s] with n = h*HD+d  (for PV frags)
// ---------------------------------------------------------------------------
#define LDSK 40   // 32 + 8 pad (bf16 elems) to break bank conflicts
__global__ __launch_bounds__(256, 1)
void gemm_bf16_wmma(const __bf16* __restrict__ A, const __bf16* __restrict__ Bw,
                    float* __restrict__ outF, __bf16* __restrict__ outB,
                    int M, int N, int K, int mode) {
    __shared__ __bf16 As[128 * LDSK];
    __shared__ __bf16 Bs[128 * LDSK];

    const int tid  = threadIdx.x;
    const int wid  = tid >> 5;
    const int lane = tid & 31;
    const int half = lane >> 4;
    const int l16  = lane & 15;
    const int m0   = blockIdx.y * 128;
    const int n0   = blockIdx.x * 128;
    const int wm   = wid & 3;   // wave row  : 4 x 32
    const int wn   = wid >> 2;  // wave col  : 2 x 64

    v8f acc[2][4];
#pragma unroll
    for (int i = 0; i < 2; ++i)
#pragma unroll
        for (int j = 0; j < 4; ++j)
#pragma unroll
            for (int r = 0; r < 8; ++r) acc[i][j][r] = 0.0f;

    for (int k0 = 0; k0 < K; k0 += 32) {
        // cooperative 128x32 tile loads (16B per thread per pass, 2 passes each)
#pragma unroll
        for (int p = 0; p < 2; ++p) {
            int idx = p * 256 + tid;          // 0..511
            int r   = idx >> 2;               // 0..127
            int cg  = idx & 3;                // 8-elem column group
            *(v8bf*)&As[r * LDSK + cg * 8] =
                *(const v8bf*)&A[(size_t)(m0 + r) * K + k0 + cg * 8];
            *(v8bf*)&Bs[r * LDSK + cg * 8] =
                *(const v8bf*)&Bw[(size_t)(n0 + r) * K + k0 + cg * 8];
        }
        if (k0 + 32 < K) {  // gfx1250 global_prefetch_b8 on next K tile
            __builtin_prefetch(&A[(size_t)(m0 + (tid >> 1)) * K + k0 + 32], 0, 1);
            __builtin_prefetch(&Bw[(size_t)(n0 + (tid >> 1)) * K + k0 + 32], 0, 1);
        }
        __syncthreads();

        v16bf af[2], bfv[4];
#pragma unroll
        for (int i = 0; i < 2; ++i)
            af[i] = load_frag(&As[(wm * 32 + i * 16 + l16) * LDSK + half * 8]);
#pragma unroll
        for (int j = 0; j < 4; ++j)
            bfv[j] = load_frag(&Bs[(wn * 64 + j * 16 + l16) * LDSK + half * 8]);

#pragma unroll
        for (int i = 0; i < 2; ++i)
#pragma unroll
            for (int j = 0; j < 4; ++j)
                acc[i][j] = __builtin_amdgcn_wmma_f32_16x16x32_bf16(
                    false, af[i], false, bfv[j], (short)0, acc[i][j], false, false);
        __syncthreads();
    }

    // epilogue — C layout: vgpr r, lanes0-15 -> M=r, lanes16-31 -> M=r+8; N = lane&15
#pragma unroll
    for (int i = 0; i < 2; ++i)
#pragma unroll
        for (int j = 0; j < 4; ++j)
#pragma unroll
            for (int r = 0; r < 8; ++r) {
                int gm = m0 + wm * 32 + i * 16 + r + 8 * half;
                int gn = n0 + wn * 64 + j * 16 + l16;
                float v = acc[i][j][r];
                if (mode == 1) {
                    outF[(size_t)gm * N + gn] = v;
                } else if (mode == 0) {
                    outB[(size_t)gm * N + gn] = (__bf16)v;
                } else {  // mode 2: V transposed [b][n][s]
                    int b = gm >> 11;          // / SEQ
                    int s = gm & (SEQ - 1);
                    outB[(size_t)b * HIDN * SEQ + (size_t)gn * SEQ + s] = (__bf16)v;
                }
            }
}

// ---------------------------------------------------------------------------
// fused per-head RMSNorm + RoPE (in-place on bf16 Q and K). One wave per row.
// ---------------------------------------------------------------------------
__global__ __launch_bounds__(256)
void qk_norm_rope(__bf16* __restrict__ Qb, __bf16* __restrict__ Kb,
                  const float* __restrict__ qw, const float* __restrict__ kw) {
    const int wid  = threadIdx.x >> 5;
    const int lane = threadIdx.x & 31;
    const size_t rid = (size_t)blockIdx.x * 8 + wid;   // (b*SEQ+s)*NH + h
    const int s = (int)((rid / NH) % SEQ);

    __bf16* rows[2] = {Qb + rid * HD, Kb + rid * HD};
    const float* wts[2] = {qw, kw};

    const int da = lane, db = lane + 32;   // rotary pair indices 0..63
    const float lg = __log2f(THETA);
    const float fa = exp2f(-((float)(2 * da) / (float)HD) * lg);
    const float fb = exp2f(-((float)(2 * db) / (float)HD) * lg);
    float ca, sa, cb, sb;
    __sincosf((float)s * fa, &sa, &ca);
    __sincosf((float)s * fb, &sb, &cb);

#pragma unroll
    for (int which = 0; which < 2; ++which) {
        __bf16* row = rows[which];
        const float* w = wts[which];
        float x1a = (float)row[da],     x2a = (float)row[da + 64];
        float x1b = (float)row[db],     x2b = (float)row[db + 64];
        float ss = x1a * x1a + x2a * x2a + x1b * x1b + x2b * x2b;
#pragma unroll
        for (int m = 1; m < 32; m <<= 1) ss += __shfl_xor(ss, m, 32);
        float rr = rsqrtf(ss * (1.0f / (float)HD) + EPSV);
        x1a *= rr * w[da];  x2a *= rr * w[da + 64];
        x1b *= rr * w[db];  x2b *= rr * w[db + 64];
        row[da]      = (__bf16)(x1a * ca - x2a * sa);
        row[da + 64] = (__bf16)(x2a * ca + x1a * sa);
        row[db]      = (__bf16)(x1b * cb - x2b * sb);
        row[db + 64] = (__bf16)(x2b * cb + x1b * sb);
    }
}

// ---------------------------------------------------------------------------
// sliding-window flash attention. One wave per 16-query tile, 8 waves/block.
// Per 32-key chunk: 8 WMMAs (QK^T over D=128) + online softmax + 8 WMMAs (PV).
// ---------------------------------------------------------------------------
__global__ __launch_bounds__(256, 1)
void attn_swa(const __bf16* __restrict__ Qb, const __bf16* __restrict__ Kb,
              const __bf16* __restrict__ Vt, __bf16* __restrict__ Ob) {
    __shared__ __bf16 pbuf[8][16 * 32];   // per-wave P repack scratch (C->A layout)

    const int wid  = threadIdx.x >> 5;
    const int lane = threadIdx.x & 31;
    const int half = lane >> 4;
    const int l16  = lane & 15;

    const int tile = blockIdx.x * 8 + wid;      // 0 .. B*NH*(SEQ/16)-1
    const int qt   = tile & (SEQ / 16 - 1);
    const int bh   = tile >> 7;                 // SEQ/16 == 128
    const int hh   = bh & (NH - 1);
    const int bb   = bh >> 4;                   // NH == 16
    const int q0   = qt * 16;
    const float scale = 0.08838834764831845f;   // 1/sqrt(128)

    // Q fragments: 4 K-blocks of 32 over D=128
    v16bf qf[4];
    {
        const __bf16* qbase = Qb + ((size_t)(bb * SEQ + q0 + l16) * NH + hh) * HD;
#pragma unroll
        for (int kb = 0; kb < 4; ++kb)
            qf[kb] = load_frag(qbase + kb * 32 + half * 8);
    }

    v8f oacc[8];
    float mrow[8], lrow[8];
#pragma unroll
    for (int t = 0; t < 8; ++t)
#pragma unroll
        for (int r = 0; r < 8; ++r) oacc[t][r] = 0.0f;
#pragma unroll
    for (int r = 0; r < 8; ++r) { mrow[r] = -1e30f; lrow[r] = 0.0f; }

    int lo_k = q0 - (WIN - 1);
    if (lo_k < 0) lo_k = 0;
    lo_k &= ~31;

    const __bf16* vbase = Vt + (size_t)bh * HD * SEQ;

    for (int k0 = lo_k; k0 <= q0 + 15; k0 += 32) {
        // ---- S = Q K^T (two 16-key N tiles) ----
        v8f sacc[2];
#pragma unroll
        for (int t = 0; t < 2; ++t)
#pragma unroll
            for (int r = 0; r < 8; ++r) sacc[t][r] = 0.0f;
#pragma unroll
        for (int t = 0; t < 2; ++t) {
#pragma unroll
            for (int kb = 0; kb < 4; ++kb) {
                const __bf16* kp =
                    Kb + ((size_t)(bb * SEQ + k0 + t * 16 + l16) * NH + hh) * HD +
                    kb * 32 + half * 8;
                v16bf kf = load_frag(kp);
                sacc[t] = __builtin_amdgcn_wmma_f32_16x16x32_bf16(
                    false, qf[kb], false, kf, (short)0, sacc[t], false, false);
            }
        }
        // ---- scale + sliding-window causal mask ----
#pragma unroll
        for (int t = 0; t < 2; ++t)
#pragma unroll
            for (int r = 0; r < 8; ++r) {
                int i = q0 + r + 8 * half;
                int j = k0 + t * 16 + l16;
                bool ok = (j <= i) && (j > i - WIN);
                sacc[t][r] = ok ? sacc[t][r] * scale : -1e30f;
            }
        // ---- online softmax (rows live across 16-lane halves) ----
        float cf[8];
#pragma unroll
        for (int r = 0; r < 8; ++r) {
            float cm = fmaxf(sacc[0][r], sacc[1][r]);
#pragma unroll
            for (int m = 1; m < 16; m <<= 1) cm = fmaxf(cm, __shfl_xor(cm, m, 32));
            float mn = fmaxf(mrow[r], cm);
            cf[r] = __expf(mrow[r] - mn);
            mrow[r] = mn;
            float p0 = __expf(sacc[0][r] - mn);
            float p1 = __expf(sacc[1][r] - mn);
            sacc[0][r] = p0; sacc[1][r] = p1;
            float rs = p0 + p1;
#pragma unroll
            for (int m = 1; m < 16; m <<= 1) rs += __shfl_xor(rs, m, 32);
            lrow[r] = lrow[r] * cf[r] + rs;
        }
#pragma unroll
        for (int t = 0; t < 8; ++t)
#pragma unroll
            for (int r = 0; r < 8; ++r) oacc[t][r] *= cf[r];

        // ---- repack P (C layout) -> A-fragment layout through wave-private LDS ----
#pragma unroll
        for (int t = 0; t < 2; ++t)
#pragma unroll
            for (int r = 0; r < 8; ++r)
                pbuf[wid][(r + 8 * half) * 32 + t * 16 + l16] = (__bf16)sacc[t][r];
        v16bf pf = load_frag(&pbuf[wid][l16 * 32 + half * 8]);  // DS in-order within wave

        // ---- O += P V  (8 d-tiles of 16; V transposed -> contiguous key loads) ----
#pragma unroll
        for (int t = 0; t < 8; ++t) {
            const __bf16* vp = vbase + (size_t)(t * 16 + l16) * SEQ + k0 + half * 8;
            v16bf vf = load_frag(vp);
            oacc[t] = __builtin_amdgcn_wmma_f32_16x16x32_bf16(
                false, pf, false, vf, (short)0, oacc[t], false, false);
        }
    }

    // ---- epilogue: normalize rows by l, store bf16 [b,s,h,d] ----
#pragma unroll
    for (int r = 0; r < 8; ++r) {
        float inv = 1.0f / lrow[r];
        int srow = q0 + r + 8 * half;
        __bf16* op = Ob + ((size_t)(bb * SEQ + srow) * NH + hh) * HD;
#pragma unroll
        for (int t = 0; t < 8; ++t)
            op[t * 16 + l16] = (__bf16)(oacc[t][r] * inv);
    }
}

// ---------------------------------------------------------------------------
// launcher
// ---------------------------------------------------------------------------
extern "C" void kernel_launch(void* const* d_in, const int* in_sizes, int n_in,
                              void* d_out, int out_size, void* d_ws, size_t ws_size,
                              hipStream_t stream) {
    const float* x  = (const float*)d_in[0];
    const float* Wq = (const float*)d_in[1];
    const float* Wk = (const float*)d_in[2];
    const float* Wv = (const float*)d_in[3];
    const float* Wo = (const float*)d_in[4];
    const float* qw = (const float*)d_in[5];
    const float* kw = (const float*)d_in[6];
    float* out = (float*)d_out;

    char* w = (char*)d_ws;
    const size_t szX = (size_t)BS * HIDN * sizeof(__bf16);     // 16.8 MB
    const size_t szW = (size_t)HIDN * HIDN * sizeof(__bf16);   // 8.4 MB
    __bf16* Xb  = (__bf16*)w;              w += szX;
    __bf16* Wqb = (__bf16*)w;              w += szW;
    __bf16* Wkb = (__bf16*)w;              w += szW;
    __bf16* Wvb = (__bf16*)w;              w += szW;
    __bf16* Wob = (__bf16*)w;              w += szW;
    __bf16* Qb  = (__bf16*)w;              w += szX;
    __bf16* Kb  = (__bf16*)w;              w += szX;
    __bf16* Vt  = (__bf16*)w;              w += szX;
    __bf16* Ab  = (__bf16*)w;              w += szX;

    // 1) convert inputs to bf16
    {
        int nX = BS * HIDN, nW = HIDN * HIDN;
        cvt_f32_bf16<<<nX / 1024, 256, 0, stream>>>(x, Xb, nX);
        cvt_f32_bf16<<<nW / 1024, 256, 0, stream>>>(Wq, Wqb, nW);
        cvt_f32_bf16<<<nW / 1024, 256, 0, stream>>>(Wk, Wkb, nW);
        cvt_f32_bf16<<<nW / 1024, 256, 0, stream>>>(Wv, Wvb, nW);
        cvt_f32_bf16<<<nW / 1024, 256, 0, stream>>>(Wo, Wob, nW);
    }

    // 2) QKV projections (V written transposed for PV fragment loads)
    dim3 ggrid(HIDN / 128, BS / 128);
    gemm_bf16_wmma<<<ggrid, 256, 0, stream>>>(Xb, Wqb, nullptr, Qb, BS, HIDN, HIDN, 0);
    gemm_bf16_wmma<<<ggrid, 256, 0, stream>>>(Xb, Wkb, nullptr, Kb, BS, HIDN, HIDN, 0);
    gemm_bf16_wmma<<<ggrid, 256, 0, stream>>>(Xb, Wvb, nullptr, Vt, BS, HIDN, HIDN, 2);

    // 3) per-head RMSNorm + RoPE on Q and K (in place)
    qk_norm_rope<<<(BS * NH) / 8, 256, 0, stream>>>(Qb, Kb, qw, kw);

    // 4) sliding-window flash attention
    attn_swa<<<(BATCH * NH * (SEQ / 16)) / 8, 256, 0, stream>>>(Qb, Kb, Vt, Ab);

    // 5) output projection -> f32 d_out
    gemm_bf16_wmma<<<ggrid, 256, 0, stream>>>(Ab, Wob, out, nullptr, BS, HIDN, HIDN, 1);
}
